// CausalSelfAttention_16200616640956
// MI455X (gfx1250) — matrix operand order repaired and checked by reading, not compiled, not run on hardware
//
#include <hip/hip_runtime.h>

#define BB 2
#define TT 2048
#define DD 1024
#define HH 16
#define DH 64

typedef unsigned short u16;
typedef __attribute__((ext_vector_type(8)))  float  v8f;
typedef __attribute__((ext_vector_type(8)))  __bf16 v8bf;
typedef __attribute__((ext_vector_type(16))) __bf16 v16bf;
typedef __attribute__((ext_vector_type(4)))  unsigned int u32x4;
typedef __attribute__((ext_vector_type(4)))  int i32x4;
typedef __attribute__((ext_vector_type(8)))  int i32x8;

static __device__ __forceinline__ u16 f2bf(float f) {
  unsigned int u = __float_as_uint(f);
  u = u + 0x7FFFu + ((u >> 16) & 1u);           // round-to-nearest-even
  return (u16)(u >> 16);
}

// Build a 16-element bf16 fragment from two 16-byte chunks.
static __device__ __forceinline__ v16bf ldfrag(const u16* p0, const u16* p1) {
  union { uint4 u; v8bf v; } a, b;
  a.u = *(const uint4*)p0;
  b.u = *(const uint4*)p1;
  return __builtin_shufflevector(a.v, b.v, 0,1,2,3,4,5,6,7,8,9,10,11,12,13,14,15);
}

// B-fragment of all 1.0bf16 (for rowsum-via-WMMA).
static __device__ __forceinline__ v16bf ones_frag() {
  union { unsigned int u[8]; v16bf v; } t;
#pragma unroll
  for (int i = 0; i < 8; ++i) t.u[i] = 0x3F803F80u;
  return t.v;
}

static __device__ __forceinline__ v8f wmma_bf16(v16bf a, v16bf b, v8f c) {
  return __builtin_amdgcn_wmma_f32_16x16x32_bf16(false, a, false, b, (short)0, c,
                                                 false, false);
}

// ---- Tensor Data Mover: DMA a 2-D bf16 tile (tile_w x tile_h, row stride
// row_stride elems) from global into contiguous LDS at lds_off.  D# per CDNA5
// ISA 8.3/8.4: group0 = {count=1, lds_addr, global_addr, type=2},
// group1 = {data_size=2B, tensor_dim0/1, tile_dim0/1, tensor_dim0_stride}.
static __device__ __forceinline__ void tdm_load_tile_2d(
    unsigned lds_off, const u16* g, unsigned tensor_w, unsigned tile_w,
    unsigned tile_h, unsigned row_stride) {
#if defined(__HIP_DEVICE_COMPILE__)
  unsigned long long ga = (unsigned long long)(size_t)g;
  u32x4 g0 = { 1u,                                     // count=1 (valid D#)
               lds_off,                                // lds_addr (bytes)
               (unsigned)ga,                           // global_addr[31:0]
               (unsigned)((ga >> 32) & 0x01FFFFFFull) | (2u << 30) };  // type=2
  const unsigned td1 = 1u << 20;                       // tensor_dim1 (ample)
  i32x8 g1 = { (int)(1u << 16),                        // data_size=1 -> 2 bytes
               (int)((tensor_w & 0xFFFFu) << 16),      // tensor_dim0 lo16
               (int)(((tensor_w >> 16) & 0xFFFFu) | ((td1 & 0xFFFFu) << 16)),
               (int)(((td1 >> 16) & 0xFFFFu) | ((tile_w & 0xFFFFu) << 16)),
               (int)(tile_h & 0xFFFFu),                // tile_dim1 (tile_dim2=0)
               (int)row_stride,                        // tensor_dim0_stride lo32
               0, 0 };
  i32x4 z4 = {};
#if __clang_major__ >= 23
  i32x8 z8 = {};
  __builtin_amdgcn_tensor_load_to_lds(g0, g1, z4, z4, z8, 0);
#else
  __builtin_amdgcn_tensor_load_to_lds(g0, g1, z4, z4, 0);
#endif
#else
  (void)lds_off; (void)g; (void)tensor_w; (void)tile_w; (void)tile_h;
  (void)row_stride;
#endif
}

static __device__ __forceinline__ void wait_tensor0() {
#if defined(__HIP_DEVICE_COMPILE__)
  __builtin_amdgcn_s_wait_tensorcnt(0);
#endif
}

static __device__ __forceinline__ unsigned lds_off_of(const void* p) {
  return (unsigned)(size_t)p;  // LDS aperture: addr[31:0] is the LDS offset
}

// ---------------- pre-pass: fp32 -> bf16 ----------------
__global__ void k_f32_to_bf16(const float* __restrict__ in, u16* __restrict__ out,
                              int n) {
  int i = (blockIdx.x * blockDim.x + threadIdx.x) * 4;
  if (i + 3 < n) {
    float4 v = *(const float4*)(in + i);
    out[i + 0] = f2bf(v.x);
    out[i + 1] = f2bf(v.y);
    out[i + 2] = f2bf(v.z);
    out[i + 3] = f2bf(v.w);
  }
}

// w[K][N] fp32 -> wT[N][K] bf16  (so GEMM B-fragments are contiguous in K)
__global__ void k_transpose_bf16(const float* __restrict__ w, u16* __restrict__ wT,
                                 int K, int N) {
  long long i = (long long)blockIdx.x * blockDim.x + threadIdx.x;
  if (i < (long long)K * N) {
    int n = (int)(i / K);
    int k = (int)(i % K);
    wT[i] = f2bf(w[(long long)k * N + n]);
  }
}

// ---------------- tiled bf16 GEMM:  C[128x128] tile per block ----------------
// A: [M][K] bf16 row-major.  Bt: [N][K] bf16 (pre-transposed weights).
// Tiles are staged into double-buffered LDS by the Tensor Data Mover (wave 0
// DMAs the A tile, wave 1 the B tile; next tile's DMA is issued before the
// current tile's WMMAs, closed by s_wait_tensorcnt + barrier).
// 8 waves: wave = (wm 0..3 row strip of 32) x (wn 0..1 col strip of 64).
// MODE 0: QKV epilogue -> Q (pre-scaled by 1/sqrt(DH)), K to [B,H,T,DH],
//         V transposed to [B,H,DH,T] (all bf16)
// MODE 1: fp32 output + bias
template <int MODE>
__global__ __launch_bounds__(256) void k_gemm(
    const u16* __restrict__ A, const u16* __restrict__ Bt,
    const float* __restrict__ bias, u16* __restrict__ qb, u16* __restrict__ kb,
    u16* __restrict__ vt, float* __restrict__ outf, int K) {
  __shared__ __align__(16) u16 As[2][128 * 32];
  __shared__ __align__(16) u16 Bs[2][128 * 32];
  const int tid   = threadIdx.x;
  const int lane  = tid & 31;
  const int wave  = tid >> 5;
  const int lhalf = lane >> 4;
  const int lcol  = lane & 15;
  const int wm = wave & 3;   // 32-row strip
  const int wn = wave >> 2;  // 64-col strip
  const int mb = blockIdx.x * 128;
  const int nb = blockIdx.y * 128;

  v8f acc[2][4];
#pragma unroll
  for (int rt = 0; rt < 2; ++rt)
#pragma unroll
    for (int cn = 0; cn < 4; ++cn) acc[rt][cn] = v8f{};

  const u16* abase = A + (size_t)mb * K;
  const u16* bbase = Bt + (size_t)nb * K;

  // prologue: DMA tile 0 into buffer 0
  if (wave == 0)      tdm_load_tile_2d(lds_off_of(&As[0][0]), abase, K, 32, 128, K);
  else if (wave == 1) tdm_load_tile_2d(lds_off_of(&Bs[0][0]), bbase, K, 32, 128, K);
  wait_tensor0();
  __syncthreads();

  int cur = 0;
  for (int kk = 0; kk < K; kk += 32) {
    const int nxt = cur ^ 1;
    if (kk + 32 < K) {  // issue next tile's DMA before computing this one
      if (wave == 0)
        tdm_load_tile_2d(lds_off_of(&As[nxt][0]), abase + kk + 32, K, 32, 128, K);
      else if (wave == 1)
        tdm_load_tile_2d(lds_off_of(&Bs[nxt][0]), bbase + kk + 32, K, 32, 128, K);
    }

    const int ar0 = wm * 32 + lcol;
    v16bf af0 = ldfrag(&As[cur][ar0 * 32 + 8 * lhalf],
                       &As[cur][ar0 * 32 + 8 * lhalf + 16]);
    v16bf af1 = ldfrag(&As[cur][(ar0 + 16) * 32 + 8 * lhalf],
                       &As[cur][(ar0 + 16) * 32 + 8 * lhalf + 16]);
    v16bf bfr[4];
#pragma unroll
    for (int cn = 0; cn < 4; ++cn) {
      const int br = wn * 64 + cn * 16 + lcol;
      bfr[cn] = ldfrag(&Bs[cur][br * 32 + 8 * lhalf],
                       &Bs[cur][br * 32 + 8 * lhalf + 16]);
    }
#pragma unroll
    for (int cn = 0; cn < 4; ++cn) {
      acc[0][cn] = wmma_bf16(af0, bfr[cn], acc[0][cn]);
      acc[1][cn] = wmma_bf16(af1, bfr[cn], acc[1][cn]);
    }

    wait_tensor0();      // issuing waves: next tile landed in LDS
    __syncthreads();     // visibility + protects buffer reuse
    cur = nxt;
  }

#pragma unroll
  for (int cn = 0; cn < 4; ++cn) {
    const int n = nb + wn * 64 + cn * 16 + lcol;
    const float bv = bias[n];
#pragma unroll
    for (int rt = 0; rt < 2; ++rt) {
#pragma unroll
      for (int idx = 0; idx < 8; ++idx) {
        const int m = mb + wm * 32 + rt * 16 + idx + 8 * lhalf;
        float val = acc[rt][cn][idx] + bv;
        if (MODE == 0) {
          const int which = n >> 10;
          const int r = n & 1023;
          const int h = r >> 6;
          const int dh = r & 63;
          const int b = m >> 11;
          const int t = m & 2047;
          if (which == 0) {  // fold softmax scale 1/sqrt(DH) into Q
            qb[((size_t)(b * HH + h) * TT + t) * DH + dh] = f2bf(val * 0.125f);
          } else if (which == 1) {
            kb[((size_t)(b * HH + h) * TT + t) * DH + dh] = f2bf(val);
          } else {
            vt[((size_t)(b * HH + h) * DH + dh) * TT + t] = f2bf(val);
          }
        } else {
          outf[(size_t)m * DD + n] = val;
        }
      }
    }
  }
}

// ---------------- flash attention ----------------
// grid: (T/128, B*H), block 256.  Each wave owns 16 q rows, key tiles of 64.
// rowsum(P) is accumulated with an extra WMMA against a ones-fragment (osum),
// so only the row-max needs cross-lane shuffles.
__global__ __launch_bounds__(256) void k_attn(const u16* __restrict__ qb,
                                              const u16* __restrict__ kb,
                                              const u16* __restrict__ vt,
                                              u16* __restrict__ aout) {
  __shared__ __align__(16) u16 pbuf[8][16 * 64];  // per-wave P tile (16x64)
  const int lane  = threadIdx.x & 31;
  const int wave  = threadIdx.x >> 5;
  const int lhalf = lane >> 4;
  const int lcol  = lane & 15;
  const int bh = blockIdx.y;  // b*H + h
  const int b  = bh >> 4;
  const int h  = bh & 15;
  const int q0 = blockIdx.x * 128 + wave * 16;

  const size_t base_t = (size_t)bh * TT;
  const u16* qrow = qb + (base_t + q0 + lcol) * DH;  // Q pre-scaled by 0.125
  const v16bf aq0 = ldfrag(qrow + 8 * lhalf, qrow + 8 * lhalf + 16);
  const v16bf aq1 = ldfrag(qrow + 32 + 8 * lhalf, qrow + 32 + 8 * lhalf + 16);
  const v16bf ones = ones_frag();

  v8f o0 = {}, o1 = {}, o2 = {}, o3 = {}, osum = {};
  float mrow[8];
#pragma unroll
  for (int i = 0; i < 8; ++i) mrow[i] = -1e30f;

  u16* pb = pbuf[wave];
  const size_t vbase = (size_t)bh * DH;
  const int ktmax = (q0 + 15) >> 6;

  for (int kt = 0; kt <= ktmax; ++kt) {
    const int kk = kt << 6;
    if (kt < ktmax) __builtin_prefetch(kb + (base_t + kk + 64 + lcol) * DH, 0, 1);

    // ---- S = Q K^T for 64 keys: batch all K-fragment loads, then 8 WMMAs ----
    const u16* kr0 = kb + (base_t + kk + 0 + lcol) * DH + 8 * lhalf;
    const u16* kr1 = kb + (base_t + kk + 16 + lcol) * DH + 8 * lhalf;
    const u16* kr2 = kb + (base_t + kk + 32 + lcol) * DH + 8 * lhalf;
    const u16* kr3 = kb + (base_t + kk + 48 + lcol) * DH + 8 * lhalf;
    v16bf kf[4][2];
    kf[0][0] = ldfrag(kr0, kr0 + 16);  kf[0][1] = ldfrag(kr0 + 32, kr0 + 48);
    kf[1][0] = ldfrag(kr1, kr1 + 16);  kf[1][1] = ldfrag(kr1 + 32, kr1 + 48);
    kf[2][0] = ldfrag(kr2, kr2 + 16);  kf[2][1] = ldfrag(kr2 + 32, kr2 + 48);
    kf[3][0] = ldfrag(kr3, kr3 + 16);  kf[3][1] = ldfrag(kr3 + 32, kr3 + 48);
    v8f s0 = {}, s1 = {}, s2 = {}, s3 = {};
    s0 = wmma_bf16(aq0, kf[0][0], s0);  s0 = wmma_bf16(aq1, kf[0][1], s0);
    s1 = wmma_bf16(aq0, kf[1][0], s1);  s1 = wmma_bf16(aq1, kf[1][1], s1);
    s2 = wmma_bf16(aq0, kf[2][0], s2);  s2 = wmma_bf16(aq1, kf[2][1], s2);
    s3 = wmma_bf16(aq0, kf[3][0], s3);  s3 = wmma_bf16(aq1, kf[3][1], s3);

    // ---- V fragments: issue loads early, they overlap the softmax VALU ----
    v16bf vf[4][2];
#pragma unroll
    for (int cn = 0; cn < 4; ++cn) {
      const u16* vr = vt + (vbase + cn * 16 + lcol) * TT + kk + 8 * lhalf;
      vf[cn][0] = ldfrag(vr, vr + 16);
      vf[cn][1] = ldfrag(vr + 32, vr + 48);
    }

    // ---- online softmax over 64 keys ----
    const int key0 = kk + lcol;
#pragma unroll
    for (int idx = 0; idx < 8; ++idx) {
      const int qr = q0 + idx + 8 * lhalf;  // global q row (C layout)
      float a0 = (key0 <= qr)      ? s0[idx] : -1e30f;
      float a1 = (key0 + 16 <= qr) ? s1[idx] : -1e30f;
      float a2 = (key0 + 32 <= qr) ? s2[idx] : -1e30f;
      float a3 = (key0 + 48 <= qr) ? s3[idx] : -1e30f;
      float mx = fmaxf(fmaxf(a0, a1), fmaxf(a2, a3));
      mx = fmaxf(mx, __shfl_xor(mx, 1, 32));
      mx = fmaxf(mx, __shfl_xor(mx, 2, 32));
      mx = fmaxf(mx, __shfl_xor(mx, 4, 32));
      mx = fmaxf(mx, __shfl_xor(mx, 8, 32));
      const float mnew = fmaxf(mrow[idx], mx);
      const float alpha = __expf(mrow[idx] - mnew);
      mrow[idx] = mnew;
      o0[idx] *= alpha;
      o1[idx] *= alpha;
      o2[idx] *= alpha;
      o3[idx] *= alpha;
      osum[idx] *= alpha;  // rowsum accumulator rescales exactly like O
      const int prow = idx + 8 * lhalf;
      pb[prow * 64 + 0 + lcol]  = f2bf(__expf(a0 - mnew));
      pb[prow * 64 + 16 + lcol] = f2bf(__expf(a1 - mnew));
      pb[prow * 64 + 32 + lcol] = f2bf(__expf(a2 - mnew));
      pb[prow * 64 + 48 + lcol] = f2bf(__expf(a3 - mnew));
    }
    // wave-private LDS round trip: C layout -> A layout
    asm volatile("s_wait_dscnt 0" ::: "memory");
    const v16bf pa0 = ldfrag(&pb[lcol * 64 + 8 * lhalf],
                             &pb[lcol * 64 + 8 * lhalf + 16]);
    const v16bf pa1 = ldfrag(&pb[lcol * 64 + 32 + 8 * lhalf],
                             &pb[lcol * 64 + 32 + 8 * lhalf + 16]);

    // ---- O += P V  (+ rowsum via ones-fragment) ----
    o0 = wmma_bf16(pa0, vf[0][0], o0);  o0 = wmma_bf16(pa1, vf[0][1], o0);
    o1 = wmma_bf16(pa0, vf[1][0], o1);  o1 = wmma_bf16(pa1, vf[1][1], o1);
    o2 = wmma_bf16(pa0, vf[2][0], o2);  o2 = wmma_bf16(pa1, vf[2][1], o2);
    o3 = wmma_bf16(pa0, vf[3][0], o3);  o3 = wmma_bf16(pa1, vf[3][1], o3);
    osum = wmma_bf16(pa0, ones, osum);
    osum = wmma_bf16(pa1, ones, osum);
  }

#pragma unroll
  for (int idx = 0; idx < 8; ++idx) {
    const float inv = 1.0f / osum[idx];  // every column of osum = rowsum
    const int qr = q0 + idx + 8 * lhalf;
    const size_t orow = ((size_t)b * TT + qr) * DD + h * DH;
    aout[orow + 0 + lcol]  = f2bf(o0[idx] * inv);
    aout[orow + 16 + lcol] = f2bf(o1[idx] * inv);
    aout[orow + 32 + lcol] = f2bf(o2[idx] * inv);
    aout[orow + 48 + lcol] = f2bf(o3[idx] * inv);
  }
}

// ---------------- launch ----------------
extern "C" void kernel_launch(void* const* d_in, const int* in_sizes, int n_in,
                              void* d_out, int out_size, void* d_ws, size_t ws_size,
                              hipStream_t stream) {
  (void)in_sizes; (void)n_in; (void)out_size; (void)ws_size;
  const float* x     = (const float*)d_in[0];
  const float* w_qkv = (const float*)d_in[1];
  const float* b_qkv = (const float*)d_in[2];
  const float* w_out = (const float*)d_in[3];
  const float* b_out = (const float*)d_in[4];
  float* out = (float*)d_out;

  char* ws = (char*)d_ws;
  const size_t SZ_X  = (size_t)BB * TT * DD * 2;       // 8 MB
  const size_t SZ_WQ = (size_t)DD * 3 * DD * 2;        // 6 MB
  const size_t SZ_WO = (size_t)DD * DD * 2;            // 2 MB
  const size_t SZ_QKV = (size_t)BB * HH * TT * DH * 2; // 8 MB each
  u16* xb    = (u16*)(ws);
  u16* wqkvT = (u16*)(ws + SZ_X);
  u16* woutT = (u16*)(ws + SZ_X + SZ_WQ);
  u16* qbuf  = (u16*)(ws + SZ_X + SZ_WQ + SZ_WO);
  u16* kbuf  = (u16*)(ws + SZ_X + SZ_WQ + SZ_WO + SZ_QKV);
  u16* vtb   = (u16*)(ws + SZ_X + SZ_WQ + SZ_WO + 2 * SZ_QKV);
  u16* aout  = (u16*)(ws + SZ_X + SZ_WQ + SZ_WO + 3 * SZ_QKV);

  const int n_x = BB * TT * DD;  // 4194304
  k_f32_to_bf16<<<n_x / 4 / 256, 256, 0, stream>>>(x, xb, n_x);
  k_transpose_bf16<<<(3 * DD * DD + 255) / 256, 256, 0, stream>>>(w_qkv, wqkvT, DD,
                                                                  3 * DD);
  k_transpose_bf16<<<(DD * DD + 255) / 256, 256, 0, stream>>>(w_out, woutT, DD, DD);

  // QKV projection: M = B*T = 4096, N = 3*D = 3072, tiles of 128x128
  k_gemm<0><<<dim3(32, 24), 256, 0, stream>>>(xb, wqkvT, b_qkv, qbuf, kbuf, vtb,
                                              nullptr, DD);
  // attention: grid (T/128, B*H)
  k_attn<<<dim3(TT / 128, BB * HH), 256, 0, stream>>>(qbuf, kbuf, vtb, aout);
  // output projection: M = 4096, N = 1024
  k_gemm<1><<<dim3(32, 8), 256, 0, stream>>>(aout, woutT, b_out, nullptr, nullptr,
                                             nullptr, out, DD);
}
